// Net_28028956573874
// MI455X (gfx1250) — compile-verified, matrix operand order checked by reading, not run on hardware
//
#include <hip/hip_runtime.h>

typedef __attribute__((ext_vector_type(16))) __bf16 v16bf;
typedef __attribute__((ext_vector_type(8)))  __bf16 v8bf;
typedef __attribute__((ext_vector_type(8)))  float  v8f;
typedef long long ll;

static constexpr int Bn  = 16;
static constexpr int Nn  = 1024;
static constexpr int Dn  = 1024;
static constexpr int NPd = 1025;                  // padded (slack row/col)
static constexpr ll  NN   = (ll)Nn * Nn;          // per-batch N*N
static constexpr ll  NDl  = (ll)Nn * Dn;          // per-batch N*D
static constexpr ll  N2K  = (ll)Nn * 2048;        // per-batch N*2D
static constexpr ll  TOTE = (ll)Bn * Nn * Dn;     // 16M elements

#define DEV __device__ __forceinline__

DEV unsigned bfbits(float f) {
    unsigned u = __builtin_bit_cast(unsigned, f);
    return (u + 0x7FFFu + ((u >> 16) & 1u)) >> 16;   // RNE f32 -> bf16
}
DEV __bf16 f2bf(float f) {
    unsigned short s = (unsigned short)bfbits(f);
    return __builtin_bit_cast(__bf16, s);
}
// Generic (flat) LDS pointer -> 32-bit LDS byte offset (low 32 bits of the
// shared-aperture flat address are the workgroup-relative LDS address).
DEV unsigned ldsoff(const void* p) {
    return (unsigned)(unsigned long long)reinterpret_cast<unsigned long long>(p);
}

// ---------------------------------------------------------------------------
// Tiled bf16 WMMA GEMM:  C[m,n] = act( sum_k A[m,k]*B[k,n] + bias[n] + addm[m,n] )
// Block tile 128x128, K-step 64, double-buffered LDS with async global->LDS
// staging (ASYNCcnt) pipelined against WMMA issue.
// 8 waves, each wave: 32x64 output (2x4 WMMA 16x16 tiles).
// TA: A stored as (K x M) row-major.  TB: B stored as (N x K) row-major.
// LDS layout is always As[m][k], Bs[n][k] so fragment loads are contiguous.
// ---------------------------------------------------------------------------
static constexpr int BM = 128, BNt = 128, BK = 64, KPAD = 72;

template <bool TA, bool TB, bool RELU, bool BIAS, bool ADDM, bool OF32, bool OBF>
__global__ __launch_bounds__(256) void gemm_bf16_k(
    const __bf16* __restrict__ Ag, ll sA, int lda,
    const __bf16* __restrict__ Bg, ll sB, int ldb,
    const float* __restrict__ bias,
    const float* __restrict__ addm, ll sAdd, int ldadd,
    float* __restrict__ Cf, ll sCf, int ldcf,
    __bf16* __restrict__ Cb, ll sCb, int ldcb, int colOff,
    int K)
{
    __shared__ __align__(16) __bf16 As[2 * BM * KPAD];
    __shared__ __align__(16) __bf16 Bs[2 * BNt * KPAD];

    const int tid  = threadIdx.x;
    const int lane = tid & 31;
    const int wv   = tid >> 5;
    const int z    = blockIdx.z;
    const int m0   = blockIdx.x * BM;
    const int n0   = blockIdx.y * BNt;

    Ag += (ll)z * sA;
    Bg += (ll)z * sB;

    const int rowW = (wv & 3) * 32;   // wave tile origin inside block
    const int colW = (wv >> 2) * 64;

    v8f acc[2][4];
#pragma unroll
    for (int i = 0; i < 2; ++i)
#pragma unroll
        for (int j = 0; j < 4; ++j)
#pragma unroll
            for (int e = 0; e < 8; ++e) acc[i][j][e] = 0.f;

    // ---- tile staging (issues async loads for contiguous paths) ----
    auto stage = [&](int k0, int buf) {
        __bf16* asb = As + buf * (BM * KPAD);
        __bf16* bsb = Bs + buf * (BNt * KPAD);
        if (!TA) {   // A row-major (M x K): contiguous k -> async straight to LDS
#pragma unroll
            for (int it = 0; it < 4; ++it) {
                int i = tid + it * 256;
                int m = i >> 3, kc = (i & 7) * 8;
                const __bf16* g = Ag + (ll)(m0 + m) * lda + k0 + kc;
                unsigned l = ldsoff(asb + m * KPAD + kc);
                asm volatile("global_load_async_to_lds_b128 %0, %1, off"
                             :: "v"(l), "v"(g) : "memory");
            }
        } else {     // A stored (K x M): contiguous m -> load + transpose-scatter
#pragma unroll
            for (int it = 0; it < 4; ++it) {
                int i = tid + it * 256;
                int k = i >> 4, mc = (i & 15) * 8;
                uint4 v = *reinterpret_cast<const uint4*>(Ag + (ll)(k0 + k) * lda + m0 + mc);
                const __bf16* e = reinterpret_cast<const __bf16*>(&v);
#pragma unroll
                for (int q = 0; q < 8; ++q) asb[(mc + q) * KPAD + k] = e[q];
            }
        }
        if (!TB) {   // B stored (K x N): contiguous n -> load + transpose-scatter
#pragma unroll
            for (int it = 0; it < 4; ++it) {
                int i = tid + it * 256;
                int k = i >> 4, nc = (i & 15) * 8;
                uint4 v = *reinterpret_cast<const uint4*>(Bg + (ll)(k0 + k) * ldb + n0 + nc);
                const __bf16* e = reinterpret_cast<const __bf16*>(&v);
#pragma unroll
                for (int q = 0; q < 8; ++q) bsb[(nc + q) * KPAD + k] = e[q];
            }
        } else {     // B stored (N x K): contiguous k -> async straight to LDS
#pragma unroll
            for (int it = 0; it < 4; ++it) {
                int i = tid + it * 256;
                int n = i >> 3, kc = (i & 7) * 8;
                const __bf16* g = Bg + (ll)(n0 + n) * ldb + k0 + kc;
                unsigned l = ldsoff(bsb + n * KPAD + kc);
                asm volatile("global_load_async_to_lds_b128 %0, %1, off"
                             :: "v"(l), "v"(g) : "memory");
            }
        }
    };

    const int nk = K / BK;
    stage(0, 0);
    asm volatile("s_wait_asynccnt 0x0" ::: "memory");
    __syncthreads();

    for (int t = 0; t < nk; ++t) {
        const int cur = t & 1;
        if (t + 1 < nk) stage((t + 1) * BK, cur ^ 1);   // prefetch next tile (async)

        const __bf16* asb = As + cur * (BM * KPAD);
        const __bf16* bsb = Bs + cur * (BNt * KPAD);
#pragma unroll
        for (int kk = 0; kk < BK; kk += 32) {
            const int koff = kk + ((lane >> 4) << 3);   // ISA bf16 A/B fragment k-mapping
            v16bf afr[2], bfr[4];
#pragma unroll
            for (int i = 0; i < 2; ++i) {
                const v8bf* p = reinterpret_cast<const v8bf*>(&asb[(rowW + 16 * i + (lane & 15)) * KPAD + koff]);
                afr[i] = __builtin_shufflevector(p[0], p[2], 0, 1, 2, 3, 4, 5, 6, 7, 8, 9, 10, 11, 12, 13, 14, 15);
            }
#pragma unroll
            for (int j = 0; j < 4; ++j) {
                const v8bf* p = reinterpret_cast<const v8bf*>(&bsb[(colW + 16 * j + (lane & 15)) * KPAD + koff]);
                bfr[j] = __builtin_shufflevector(p[0], p[2], 0, 1, 2, 3, 4, 5, 6, 7, 8, 9, 10, 11, 12, 13, 14, 15);
            }
#pragma unroll
            for (int i = 0; i < 2; ++i)
#pragma unroll
                for (int j = 0; j < 4; ++j)
                    acc[i][j] = __builtin_amdgcn_wmma_f32_16x16x32_bf16(
                        false, afr[i], false, bfr[j], (short)0, acc[i][j], false, false);
        }
        asm volatile("s_wait_asynccnt 0x0" ::: "memory");  // next tile landed in LDS
        __syncthreads();                                   // + all ds stores fenced
    }

    // ---- epilogue ----
    const int rbase = (lane >> 4) << 3;
#pragma unroll
    for (int i = 0; i < 2; ++i) {
#pragma unroll
        for (int j = 0; j < 4; ++j) {
            const int n = n0 + colW + 16 * j + (lane & 15);
            float bv = 0.f;
            if (BIAS) bv = bias[n];
#pragma unroll
            for (int r = 0; r < 8; ++r) {
                const int m = m0 + rowW + 16 * i + rbase + r;
                float v = acc[i][j][r] + bv;
                if (ADDM) v += addm[(ll)z * sAdd + (ll)m * ldadd + n];
                if (RELU) v = fmaxf(v, 0.f);
                if (OF32) Cf[(ll)z * sCf + (ll)m * ldcf + n] = v;
                if (OBF)  Cb[(ll)z * sCb + (ll)m * ldcb + n + colOff] = f2bf(v);
            }
        }
    }
}

// ---------------------------------------------------------------------------
// f32 -> bf16 conversion (vectorized x4)
// ---------------------------------------------------------------------------
__global__ __launch_bounds__(256) void conv_bf16_k(const float* __restrict__ in,
                                                   __bf16* __restrict__ out, int n4)
{
    int i = blockIdx.x * 256 + threadIdx.x;
    if (i >= n4) return;
    float4 v = reinterpret_cast<const float4*>(in)[i];
    uint2 o;
    o.x = bfbits(v.x) | (bfbits(v.y) << 16);
    o.y = bfbits(v.z) | (bfbits(v.w) << 16);
    reinterpret_cast<uint2*>(out)[i] = o;
}

// ---------------------------------------------------------------------------
// Column L1-normalize A (dim -2) fused with bf16 conversion.
// Block: 256 thr = 64 cols x 4 row-strips. Grid: (N/64, B).
// ---------------------------------------------------------------------------
__global__ __launch_bounds__(256) void colnorm_k(const float* __restrict__ A,
                                                 __bf16* __restrict__ An)
{
    const int b = blockIdx.y;
    const float* Ab = A + (ll)b * NN;
    __bf16* Ob = An + (ll)b * NN;
    const int j  = blockIdx.x * 64 + (threadIdx.x & 63);
    const int r0 = threadIdx.x >> 6;
    float s = 0.f;
    for (int r = r0; r < Nn; r += 4) s += fabsf(Ab[(ll)r * Nn + j]);
    __shared__ float sm[256];
    sm[threadIdx.x] = s;
    __syncthreads();
    if (r0 == 0) {
        float t = sm[threadIdx.x] + sm[threadIdx.x + 64] + sm[threadIdx.x + 128] + sm[threadIdx.x + 192];
        sm[threadIdx.x] = 1.f / fmaxf(t, 1e-12f);
    }
    __syncthreads();
    const float inv = sm[threadIdx.x & 63];
    for (int r = r0; r < Nn; r += 4) Ob[(ll)r * Nn + j] = f2bf(Ab[(ll)r * Nn + j] * inv);
}

// ---------------------------------------------------------------------------
// Instance-norm statistics: deterministic two-stage reduction
// ---------------------------------------------------------------------------
__global__ __launch_bounds__(256) void stats1_k(const float* __restrict__ S, float* __restrict__ red)
{
    const int b = blockIdx.y, c = blockIdx.x;
    const float* p = S + (ll)b * NN + (ll)c * 16384;
    float s = 0.f, ss = 0.f;
    for (int t = threadIdx.x; t < 16384; t += 256) { float v = p[t]; s += v; ss += v * v; }
    __shared__ float s1[256], s2[256];
    s1[threadIdx.x] = s; s2[threadIdx.x] = ss;
    __syncthreads();
    for (int st = 128; st > 0; st >>= 1) {
        if (threadIdx.x < st) { s1[threadIdx.x] += s1[threadIdx.x + st]; s2[threadIdx.x] += s2[threadIdx.x + st]; }
        __syncthreads();
    }
    if (threadIdx.x == 0) { red[(b * 64 + c) * 2] = s1[0]; red[(b * 64 + c) * 2 + 1] = s2[0]; }
}

__global__ __launch_bounds__(64) void stats2_k(const float* __restrict__ red, float* __restrict__ musig)
{
    const int b = blockIdx.x;
    __shared__ float s1[64], s2[64];
    s1[threadIdx.x] = red[(b * 64 + threadIdx.x) * 2];
    s2[threadIdx.x] = red[(b * 64 + threadIdx.x) * 2 + 1];
    __syncthreads();
    for (int st = 32; st > 0; st >>= 1) {
        if (threadIdx.x < st) { s1[threadIdx.x] += s1[threadIdx.x + st]; s2[threadIdx.x] += s2[threadIdx.x + st]; }
        __syncthreads();
    }
    if (threadIdx.x == 0) {
        const float inv = 1.f / (float)NN;
        float mu = s1[0] * inv;
        float var = s2[0] * inv - mu * mu;
        musig[b * 2]     = mu;
        musig[b * 2 + 1] = rsqrtf(var + 1e-5f);
    }
}

// ---------------------------------------------------------------------------
// Instance-norm apply + pad into (B,1025,1025) log-alpha (slack row/col = 0)
// ---------------------------------------------------------------------------
__global__ __launch_bounds__(256) void padnorm_k(const float* __restrict__ S,
                                                 float* __restrict__ LA,
                                                 const float* __restrict__ musig,
                                                 const float* __restrict__ inw,
                                                 const float* __restrict__ inb)
{
    const ll TOT = (ll)Bn * NPd * NPd;
    ll idx = (ll)blockIdx.x * 256 + threadIdx.x;
    if (idx >= TOT) return;
    int b = (int)(idx / ((ll)NPd * NPd));
    int rem = (int)(idx - (ll)b * NPd * NPd);
    int i = rem / NPd, j = rem - i * NPd;
    float v = 0.f;
    if (i < Nn && j < Nn)
        v = (S[(ll)b * NN + (ll)i * Nn + j] - musig[b * 2]) * musig[b * 2 + 1] * inw[0] + inb[0];
    LA[idx] = v;
}

// ---------------------------------------------------------------------------
// Sinkhorn row pass: rows 0..N-1 normalized over all 1025 cols (LDS-resident)
// ---------------------------------------------------------------------------
__global__ __launch_bounds__(256) void sk_row_k(float* __restrict__ LA)
{
    const int row = blockIdx.x, b = blockIdx.y;
    float* p = LA + ((ll)b * NPd + row) * NPd;
    __shared__ float buf[NPd];
    __shared__ float sm[256];
    float m = -3.0e38f;
    for (int j = threadIdx.x; j < NPd; j += 256) { float v = p[j]; buf[j] = v; m = fmaxf(m, v); }
    sm[threadIdx.x] = m;
    __syncthreads();
    for (int st = 128; st > 0; st >>= 1) {
        if (threadIdx.x < st) sm[threadIdx.x] = fmaxf(sm[threadIdx.x], sm[threadIdx.x + st]);
        __syncthreads();
    }
    const float rmax = sm[0];
    __syncthreads();
    float s = 0.f;
    for (int j = threadIdx.x; j < NPd; j += 256) s += __expf(buf[j] - rmax);
    sm[threadIdx.x] = s;
    __syncthreads();
    for (int st = 128; st > 0; st >>= 1) {
        if (threadIdx.x < st) sm[threadIdx.x] += sm[threadIdx.x + st];
        __syncthreads();
    }
    const float lse = rmax + __logf(sm[0]);
    for (int j = threadIdx.x; j < NPd; j += 256) p[j] = buf[j] - lse;
}

// ---------------------------------------------------------------------------
// Sinkhorn col pass: cols 0..N-1 normalized over all 1025 rows.
// Block = 8 cols x 32 row-threads, tile cached in LDS (1R+1W to memory).
// ---------------------------------------------------------------------------
__global__ __launch_bounds__(256) void sk_col_k(float* __restrict__ LA)
{
    const int b  = blockIdx.y;
    const int tx = threadIdx.x & 7;
    const int ty = threadIdx.x >> 3;
    const int c  = blockIdx.x * 8 + tx;
    __shared__ float buf[NPd * 8];
    __shared__ float sm[256];
    float m = -3.0e38f;
    for (int r = ty; r < NPd; r += 32) {
        float v = LA[((ll)b * NPd + r) * NPd + c];
        buf[r * 8 + tx] = v;
        m = fmaxf(m, v);
    }
    sm[threadIdx.x] = m;
    __syncthreads();
    for (int st = 16; st > 0; st >>= 1) {
        if (ty < st) sm[threadIdx.x] = fmaxf(sm[threadIdx.x], sm[threadIdx.x + st * 8]);
        __syncthreads();
    }
    const float cmax = sm[tx];
    __syncthreads();
    float s = 0.f;
    for (int r = ty; r < NPd; r += 32) s += __expf(buf[r * 8 + tx] - cmax);
    sm[threadIdx.x] = s;
    __syncthreads();
    for (int st = 16; st > 0; st >>= 1) {
        if (ty < st) sm[threadIdx.x] += sm[threadIdx.x + st * 8];
        __syncthreads();
    }
    const float lse = cmax + __logf(sm[tx]);
    for (int r = ty; r < NPd; r += 32) LA[((ll)b * NPd + r) * NPd + c] = buf[r * 8 + tx] - lse;
}

// ---------------------------------------------------------------------------
// exp(log_alpha interior) -> bf16 matching matrix (layer-0 cross-graph input)
// ---------------------------------------------------------------------------
__global__ __launch_bounds__(256) void expbf_k(const float* __restrict__ LA, __bf16* __restrict__ Sb)
{
    ll idx = (ll)blockIdx.x * 256 + threadIdx.x;
    if (idx >= (ll)Bn * NN) return;
    int b = (int)(idx >> 20);
    int rem = (int)(idx & (NN - 1));
    int i = rem >> 10, j = rem & 1023;
    Sb[idx] = f2bf(__expf(LA[((ll)b * NPd + i) * NPd + j]));
}

// ---------------------------------------------------------------------------
// Final: s = exp(la), srcinlier = row sums; refinlier = col sums (2 kernels)
// ---------------------------------------------------------------------------
__global__ __launch_bounds__(256) void final_row_k(const float* __restrict__ LA,
                                                   float* __restrict__ outS,
                                                   float* __restrict__ outSrc)
{
    const int row = blockIdx.x, b = blockIdx.y;
    const float* p = LA + ((ll)b * NPd + row) * NPd;
    float* o = outS + ((ll)b * NN + (ll)row * Nn);
    float s = 0.f;
    for (int j = threadIdx.x; j < Nn; j += 256) { float e = __expf(p[j]); o[j] = e; s += e; }
    __shared__ float sm[256];
    sm[threadIdx.x] = s;
    __syncthreads();
    for (int st = 128; st > 0; st >>= 1) {
        if (threadIdx.x < st) sm[threadIdx.x] += sm[threadIdx.x + st];
        __syncthreads();
    }
    if (threadIdx.x == 0) outSrc[(ll)b * Nn + row] = sm[0];
}

__global__ __launch_bounds__(256) void final_col_k(const float* __restrict__ LA,
                                                   float* __restrict__ outRef)
{
    const int b  = blockIdx.y;
    const int tx = threadIdx.x & 15, ty = threadIdx.x >> 4;
    const int c  = blockIdx.x * 16 + tx;
    float s = 0.f;
    for (int r = ty; r < Nn; r += 16) s += __expf(LA[((ll)b * NPd + r) * NPd + c]);
    __shared__ float sm[256];
    sm[threadIdx.x] = s;
    __syncthreads();
    for (int st = 8; st > 0; st >>= 1) {
        if (ty < st) sm[threadIdx.x] += sm[threadIdx.x + st * 16];
        __syncthreads();
    }
    if (ty == 0) outRef[(ll)b * Nn + c] = sm[tx];
}

// ---------------------------------------------------------------------------
// Host orchestration
// ---------------------------------------------------------------------------
extern "C" void kernel_launch(void* const* d_in, const int* in_sizes, int n_in,
                              void* d_out, int out_size, void* d_ws, size_t ws_size,
                              hipStream_t stream)
{
    const float* emb_src = (const float*)d_in[0];
    const float* emb_tgt = (const float*)d_in[1];
    const float* A_src   = (const float*)d_in[2];
    const float* A_tgt   = (const float*)d_in[3];
    const float* cgW     = (const float*)d_in[11];
    const float* cgb     = (const float*)d_in[12];

    char* w = (char*)d_ws;
    auto take = [&](size_t bytes) -> char* {
        char* r = w;
        w += (bytes + 255) & ~(size_t)255;
        return r;
    };
    __bf16* Xs   = (__bf16*)take(TOTE * 2);
    __bf16* Xt   = (__bf16*)take(TOTE * 2);
    __bf16* W1b  = (__bf16*)take((ll)Dn * Dn * 2);
    __bf16* W2b  = (__bf16*)take((ll)Dn * Dn * 2);
    __bf16* W3b  = (__bf16*)take((ll)Dn * Dn * 2);
    __bf16* CGWb = (__bf16*)take((ll)2048 * Dn * 2);
    __bf16* AnS  = (__bf16*)take(TOTE * 2);
    __bf16* AnT  = (__bf16*)take(TOTE * 2);
    __bf16* AXb  = (__bf16*)take(TOTE * 2);
    float*  UX   = (float*)take(TOTE * 4);        // reused as SRAW after GEMM3s
    __bf16* CATs = (__bf16*)take((ll)Bn * N2K * 2);
    __bf16* CATt = (__bf16*)take((ll)Bn * N2K * 2);
    __bf16* TMPb = (__bf16*)take(TOTE * 2);
    float*  LA   = (float*)take((ll)Bn * NPd * NPd * 4);
    __bf16* Sb   = (__bf16*)take(TOTE * 2);
    float*  RED  = (float*)take((ll)Bn * 64 * 2 * 4);
    float*  MUSIG= (float*)take((ll)Bn * 2 * 4);
    float*  SRAW = UX;

    float* outS   = (float*)d_out;
    float* outSrc = outS + (ll)Bn * NN;
    float* outRef = outSrc + (ll)Bn * Nn;

    const dim3 blk(256);
    const dim3 gFlat(128, 8, 1);   // M=16384, N=1024
    const dim3 gBat(8, 8, Bn);     // per-batch 1024x1024
    const int  g16M4 = (int)(TOTE / 4 / 256);      // convert grids
    const int  g1M4  = (int)((ll)Dn * Dn / 4 / 256);
    const int  g2M4  = (int)((ll)2048 * Dn / 4 / 256);

    // ---- prep ----
    conv_bf16_k<<<g16M4, blk, 0, stream>>>(emb_src, Xs, (int)(TOTE / 4));
    conv_bf16_k<<<g16M4, blk, 0, stream>>>(emb_tgt, Xt, (int)(TOTE / 4));
    conv_bf16_k<<<g2M4, blk, 0, stream>>>(cgW, CGWb, (int)((ll)2048 * Dn / 4));
    colnorm_k<<<dim3(16, Bn), blk, 0, stream>>>(A_src, AnS);
    colnorm_k<<<dim3(16, Bn), blk, 0, stream>>>(A_tgt, AnT);

    for (int L = 0; L < 2; ++L) {
        const float* Wa  = (const float*)d_in[L ? 13 : 4];
        const float* ba  = (const float*)d_in[L ? 14 : 5];
        const float* Wu  = (const float*)d_in[L ? 15 : 6];
        const float* bu  = (const float*)d_in[L ? 16 : 7];
        const float* Aff = (const float*)d_in[L ? 17 : 8];
        const float* inw = (const float*)d_in[L ? 18 : 9];
        const float* inb = (const float*)d_in[L ? 19 : 10];

        conv_bf16_k<<<g1M4, blk, 0, stream>>>(Wa, W1b, (int)((ll)Dn * Dn / 4));
        conv_bf16_k<<<g1M4, blk, 0, stream>>>(Wu, W2b, (int)((ll)Dn * Dn / 4));
        conv_bf16_k<<<g1M4, blk, 0, stream>>>(Aff, W3b, (int)((ll)Dn * Dn / 4));

        for (int g = 0; g < 2; ++g) {
            const __bf16* X  = g ? Xt : Xs;
            const __bf16* An = g ? AnT : AnS;
            __bf16* CAT = g ? CATt : CATs;
            // AX = relu(X @ Wa + ba)  -> bf16
            gemm_bf16_k<false, false, true, true, false, false, true><<<gFlat, blk, 0, stream>>>(
                X, 0, Dn, W1b, 0, Dn, ba, nullptr, 0, 0,
                nullptr, 0, 0, AXb, 0, Dn, 0, Dn);
            // UX = relu(X @ Wu + bu)  -> f32
            gemm_bf16_k<false, false, true, true, false, true, false><<<gFlat, blk, 0, stream>>>(
                X, 0, Dn, W2b, 0, Dn, bu, nullptr, 0, 0,
                UX, 0, Dn, nullptr, 0, 0, 0, Dn);
            // E = An @ AX + UX -> bf16, written into CAT left half (stride 2048)
            gemm_bf16_k<false, false, false, false, true, false, true><<<gBat, blk, 0, stream>>>(
                An, NN, Nn, AXb, NDl, Dn, nullptr, UX, NDl, Dn,
                nullptr, 0, 0, CAT, N2K, 2048, 0, Nn);
        }

        // TMP = E_s @ Aff  -> bf16   (E_s = CATs left half, lda=2048)
        gemm_bf16_k<false, false, false, false, false, false, true><<<gFlat, blk, 0, stream>>>(
            CATs, 0, 2048, W3b, 0, Dn, nullptr, nullptr, 0, 0,
            nullptr, 0, 0, TMPb, 0, Dn, 0, Dn);
        // SRAW = TMP @ E_t^T  -> f32  (TRANS_B against CATt left half)
        gemm_bf16_k<false, true, false, false, false, true, false><<<gBat, blk, 0, stream>>>(
            TMPb, NDl, Dn, CATt, N2K, 2048, nullptr, nullptr, 0, 0,
            SRAW, NN, Nn, nullptr, 0, 0, 0, Dn);

        // instance-norm stats + padded log-alpha
        stats1_k<<<dim3(64, Bn), blk, 0, stream>>>(SRAW, RED);
        stats2_k<<<Bn, 64, 0, stream>>>(RED, MUSIG);
        {
            const ll TOT = (ll)Bn * NPd * NPd;
            padnorm_k<<<(int)((TOT + 255) / 256), blk, 0, stream>>>(SRAW, LA, MUSIG, inw, inb);
        }
        // sinkhorn (slack variant): 20 x (row-normalize, col-normalize)
        for (int it = 0; it < 20; ++it) {
            sk_row_k<<<dim3(Nn, Bn), blk, 0, stream>>>(LA);
            sk_col_k<<<dim3(Nn / 8, Bn), blk, 0, stream>>>(LA);
        }

        if (L == 0) {
            expbf_k<<<(int)(TOTE / 256), blk, 0, stream>>>(LA, Sb);
            // cross_s = s @ e_t   -> CATs right half
            gemm_bf16_k<false, false, false, false, false, false, true><<<gBat, blk, 0, stream>>>(
                Sb, NN, Nn, CATt, N2K, 2048, nullptr, nullptr, 0, 0,
                nullptr, 0, 0, CATs, N2K, 2048, 1024, Nn);
            // cross_t = s^T @ e_s -> CATt right half   (TRANS_A)
            gemm_bf16_k<true, false, false, false, false, false, true><<<gBat, blk, 0, stream>>>(
                Sb, NN, Nn, CATs, N2K, 2048, nullptr, nullptr, 0, 0,
                nullptr, 0, 0, CATt, N2K, 2048, 1024, Nn);
            // new embeddings: [e, cross] @ cgW + cgb  -> bf16 inputs for layer 1
            gemm_bf16_k<false, false, false, true, false, false, true><<<gFlat, blk, 0, stream>>>(
                CATs, 0, 2048, CGWb, 0, Dn, cgb, nullptr, 0, 0,
                nullptr, 0, 0, Xs, 0, Dn, 0, 2048);
            gemm_bf16_k<false, false, false, true, false, false, true><<<gFlat, blk, 0, stream>>>(
                CATt, 0, 2048, CGWb, 0, Dn, cgb, nullptr, 0, 0,
                nullptr, 0, 0, Xt, 0, Dn, 0, 2048);
        } else {
            final_row_k<<<dim3(Nn, Bn), blk, 0, stream>>>(LA, outS, outSrc);
            final_col_k<<<dim3(Nn / 16, Bn), blk, 0, stream>>>(LA, outRef);
        }
    }
    (void)in_sizes; (void)n_in; (void)out_size; (void)ws_size;
}
